// BroadcastAttention_44229573214345
// MI455X (gfx1250) — matrix operand order repaired and checked by reading, not compiled, not run on hardware
//
#include <hip/hip_runtime.h>
#include <hip/hip_bf16.h>

// ---------------------------------------------------------------------------
// BroadcastAttention for MI455X (gfx1250).
//   Pipeline:
//     k0a: Wqkv,Wp f32 -> f16 (Wqkv zero-padded to 1088 rows)
//     k0b: x (16,512,4096) f32 -> xT (16,4096,512) f16 via TDM tensor_load_to_lds
//     k1 : qkv = Wqkv @ x + bqkv        (LDS-free f16 WMMA, direct global frags)
//     k2 : softmax(q) row -> ctx = k @ p
//     k3 : out = Wp @ (relu(v)*ctx) + bp (WMMA, B staged in LDS with fused epilogue)
// ---------------------------------------------------------------------------

typedef __attribute__((ext_vector_type(16))) _Float16 v16h;
typedef __attribute__((ext_vector_type(8)))  _Float16 v8h;
typedef __attribute__((ext_vector_type(8)))  float    v8f;
typedef __attribute__((ext_vector_type(4)))  unsigned int v4u;
typedef __attribute__((ext_vector_type(8)))  int      v8i;
typedef __attribute__((ext_vector_type(4)))  int      v4i;

#define DIMC     512
#define NHEAD    8
#define HDIM     64
#define QKV_ROWS 1032      // 8 * (1 + 2*64)
#define QKV_PAD  1088      // 17 * 64, zero-padded weight rows
#define HROW     129       // rows per head in qkv
#define NSEQ     4096
#define NBATCH   16

#define LDB 40             // padded LDS pitch (halves) -> conflict-free 16B reads

// --- WMMA fragments straight from global f16 (contiguous k order, same in A and B) -----------
// A (16x32): lane m = L%16, g = L/16; 16 contiguous k at g*16.
__device__ __forceinline__ v16h fragA_g(const _Float16* __restrict__ W,
                                        int row0, int kk, int lane) {
  const int m = lane & 15, g = lane >> 4;
  const _Float16* p = W + (size_t)(row0 + m) * DIMC + kk;
  union { v16h v; v8h h[2]; } u;
  u.h[0] = *(const v8h*)(p + g * 16);
  u.h[1] = *(const v8h*)(p + g * 16 + 8);
  return u.v;
}
// B (32x16): lane n = L%16; 16 contiguous k at g*16 (xT is c-contiguous).
__device__ __forceinline__ v16h fragB_g(const _Float16* __restrict__ xT,
                                        int nrow0, int kk, int lane) {
  const int n = lane & 15, g = lane >> 4;
  const _Float16* p = xT + (size_t)(nrow0 + n) * DIMC + kk + g * 16;
  union { v16h v; v8h h[2]; } u;
  u.h[0] = *(const v8h*)(p);
  u.h[1] = *(const v8h*)(p + 8);
  return u.v;
}
// B fragment from n-major LDS tile (pitch LDB)
__device__ __forceinline__ v16h fragB_lds(const _Float16* base, int row0, int lane) {
  const int n = lane & 15, g = lane >> 4;
  const _Float16* p = base + (size_t)(row0 + n) * LDB + g * 16;
  union { v16h v; v8h h[2]; } u;
  u.h[0] = *(const v8h*)(p);
  u.h[1] = *(const v8h*)(p + 8);
  return u.v;
}

// ---------------------------------------------------------------------------
// k0a: weight f32 -> f16, zero-pad rows >= rows_src
// ---------------------------------------------------------------------------
__global__ void __launch_bounds__(256)
wconv_kernel(const float* __restrict__ W, _Float16* __restrict__ Wh,
             int rows_src, int total) {
  const int i = blockIdx.x * 256 + threadIdx.x;
  if (i >= total) return;
  const int o = i / DIMC;
  Wh[i] = (o < rows_src) ? (_Float16)W[i] : (_Float16)0.f;
}

// ---------------------------------------------------------------------------
// k0b: x -> xT (f16, c contiguous). 64x64 f32 tile fetched by the TDM.
// ---------------------------------------------------------------------------
__global__ void __launch_bounds__(256)
x_transpose_kernel(const float* __restrict__ X, _Float16* __restrict__ xTh) {
  __shared__ __align__(16) float tile[64 * 64];   // sole LDS object
  const int b = blockIdx.z, c0 = blockIdx.y * 64, n0 = blockIdx.x * 64;
  const int t = threadIdx.x;

  if (t < 32) {  // wave 0 issues one TDM op (SGPR descriptor, EXEC-independent)
    const size_t ga = (size_t)(const void*)(X + ((size_t)b * DIMC + c0) * NSEQ + n0);
    v4u g0;
    g0.x = 1u;                                        // count=1, user descriptor
    g0.y = (unsigned int)(size_t)(&tile[0]);          // flat addr low 32b == LDS offset
    g0.z = (unsigned int)ga;                          // global_addr[31:0]
    g0.w = ((unsigned int)(ga >> 32) & 0x01FFFFFFu)   // global_addr[56:32]
           | (2u << 30);                              // type = 2 (image/TDM)
    const v8i g1 = { (int)(2u << 16),      // data_size = 2 (4 bytes)
                     (int)(NSEQ << 16),    // tensor_dim0[15:0]   = 4096
                     (int)(DIMC << 16),    // tensor_dim1[15:0]   = 512
                     (int)(64u  << 16),    // tile_dim0           = 64 (n)
                     64,                   // tile_dim1           = 64 (c rows)
                     NSEQ,                 // tensor_dim0_stride  = 4096
                     0, 0 };
    const v4i gz4 = { 0, 0, 0, 0 };
    const v8i gz8 = { 0, 0, 0, 0, 0, 0, 0, 0 };
    __builtin_amdgcn_tensor_load_to_lds(g0, g1, gz4, gz4, gz8, 0);
    __builtin_amdgcn_s_wait_tensorcnt(0);
  }
  __syncthreads();

  // transpose out: thread t -> output row n = t>>2, 16 c's at cg
  const int n = t >> 2, cg = (t & 3) * 16;
  v8h o0, o1;
#pragma unroll
  for (int j = 0; j < 8; ++j) o0[j] = (_Float16)tile[(cg + j) * 64 + n];
#pragma unroll
  for (int j = 0; j < 8; ++j) o1[j] = (_Float16)tile[(cg + 8 + j) * 64 + n];
  _Float16* dst = xTh + ((size_t)b * NSEQ + n0 + n) * DIMC + c0 + cg;
  *(v8h*)dst       = o0;
  *(v8h*)(dst + 8) = o1;
}

// ---------------------------------------------------------------------------
// k1: qkv[b,o,n] = sum_c Wh[o,c] * xT[b,n,c] + bqkv[o]  -> f16
// LDS-free: 128 threads = 4 waves, each wave a 32x64 patch (2x4 WMMA tiles).
// ---------------------------------------------------------------------------
__global__ void __launch_bounds__(128)
qkv_gemm_kernel(const _Float16* __restrict__ Wh, const _Float16* __restrict__ xTh,
                const float* __restrict__ bias, _Float16* __restrict__ qkvh) {
  const int b    = blockIdx.z;
  const int m0   = blockIdx.y * 64;
  const int n0   = blockIdx.x * 128;
  const int t    = threadIdx.x;
  const int lane = t & 31;
  const int wave = t >> 5;
  const int wr   = (wave >> 1) * 32;   // M offset of wave
  const int wc   = (wave & 1) * 64;    // N offset of wave

  const _Float16* xTb = xTh + (size_t)b * NSEQ * DIMC;

  v8f acc[2][4] = {};
#pragma unroll 4
  for (int kk = 0; kk < DIMC; kk += 32) {
    v16h a0 = fragA_g(Wh, m0 + wr,      kk, lane);
    v16h a1 = fragA_g(Wh, m0 + wr + 16, kk, lane);
    v16h b0 = fragB_g(xTb, n0 + wc,      kk, lane);
    v16h b1 = fragB_g(xTb, n0 + wc + 16, kk, lane);
    v16h b2 = fragB_g(xTb, n0 + wc + 32, kk, lane);
    v16h b3 = fragB_g(xTb, n0 + wc + 48, kk, lane);
    acc[0][0] = __builtin_amdgcn_wmma_f32_16x16x32_f16(false, a0, false, b0, (short)0, acc[0][0], false, false);
    acc[0][1] = __builtin_amdgcn_wmma_f32_16x16x32_f16(false, a0, false, b1, (short)0, acc[0][1], false, false);
    acc[0][2] = __builtin_amdgcn_wmma_f32_16x16x32_f16(false, a0, false, b2, (short)0, acc[0][2], false, false);
    acc[0][3] = __builtin_amdgcn_wmma_f32_16x16x32_f16(false, a0, false, b3, (short)0, acc[0][3], false, false);
    acc[1][0] = __builtin_amdgcn_wmma_f32_16x16x32_f16(false, a1, false, b0, (short)0, acc[1][0], false, false);
    acc[1][1] = __builtin_amdgcn_wmma_f32_16x16x32_f16(false, a1, false, b1, (short)0, acc[1][1], false, false);
    acc[1][2] = __builtin_amdgcn_wmma_f32_16x16x32_f16(false, a1, false, b2, (short)0, acc[1][2], false, false);
    acc[1][3] = __builtin_amdgcn_wmma_f32_16x16x32_f16(false, a1, false, b3, (short)0, acc[1][3], false, false);
  }

  const int g = lane >> 4, nl = lane & 15;
#pragma unroll
  for (int i = 0; i < 2; ++i)
#pragma unroll
    for (int j = 0; j < 4; ++j) {
      const int obase = m0 + wr + i * 16 + 8 * g;
      const int n     = n0 + wc + j * 16 + nl;
#pragma unroll
      for (int r = 0; r < 8; ++r) {
        const int o = obase + r;
        if (o < QKV_ROWS)
          qkvh[((size_t)b * QKV_ROWS + o) * NSEQ + n] =
              (_Float16)(acc[i][j][r] + bias[o]);
      }
    }
}

// ---------------------------------------------------------------------------
// k2: per (b,h): p = softmax(q over N); ctx[d] = sum_n k[d,n] * p[n]
// ---------------------------------------------------------------------------
__global__ void __launch_bounds__(256)
softmax_ctx_kernel(const _Float16* __restrict__ qkvh, float* __restrict__ ctx) {
  __shared__ float pbuf[NSEQ];
  __shared__ float red[256];
  const int bh = blockIdx.x, b = bh >> 3, h = bh & 7;
  const int t  = threadIdx.x;
  const _Float16* qrow = qkvh + ((size_t)b * QKV_ROWS + h * HROW) * NSEQ;

  float m = -3.4e38f;
  for (int n = t; n < NSEQ; n += 256) m = fmaxf(m, (float)qrow[n]);
  red[t] = m; __syncthreads();
  for (int s = 128; s > 0; s >>= 1) { if (t < s) red[t] = fmaxf(red[t], red[t + s]); __syncthreads(); }
  m = red[0]; __syncthreads();

  float sum = 0.f;
  for (int n = t; n < NSEQ; n += 256) {
    float e = __expf((float)qrow[n] - m);
    pbuf[n] = e;
    sum += e;
  }
  red[t] = sum; __syncthreads();
  for (int s = 128; s > 0; s >>= 1) { if (t < s) red[t] += red[t + s]; __syncthreads(); }
  const float inv = 1.f / red[0];
  __syncthreads();

  for (int d = 0; d < HDIM; ++d) {
    const _Float16* krow = qkvh + ((size_t)b * QKV_ROWS + h * HROW + 1 + d) * NSEQ;
    float s = 0.f;
    for (int n = t; n < NSEQ; n += 256) s += (float)krow[n] * pbuf[n];
    red[t] = s; __syncthreads();
    for (int st = 128; st > 0; st >>= 1) { if (t < st) red[t] += red[t + st]; __syncthreads(); }
    if (t == 0) ctx[(size_t)bh * HDIM + d] = red[0] * inv;
    __syncthreads();
  }
}

// ---------------------------------------------------------------------------
// k3: out[b,o,n] = sum_c Wp[o,c] * (relu(v[b,c,n]) * ctx[b,c]) + bp[o]
// A straight from global f16; B (relu(v)*ctx) staged n-major in LDS.
// ---------------------------------------------------------------------------
__global__ void __launch_bounds__(128)
proj_gemm_kernel(const _Float16* __restrict__ qkvh, const float* __restrict__ ctx,
                 const _Float16* __restrict__ Wph, const float* __restrict__ bias,
                 float* __restrict__ out) {
  __shared__ __align__(16) _Float16 Blds[64 * LDB];

  const int b    = blockIdx.z;
  const int m0   = blockIdx.y * 64;
  const int n0   = blockIdx.x * 64;
  const int t    = threadIdx.x;
  const int lane = t & 31;
  const int wave = t >> 5;
  const int wr   = (wave >> 1) * 32;
  const int wc   = (wave & 1) * 32;

  v8f acc[2][2] = {};

  for (int kk = 0; kk < DIMC; kk += 32) {
    if (kk) __syncthreads();
    // stage B: y[c,n] = relu(v)*ctx from qkvh (f16), transposed to n-major
    {
      const int k = t >> 2, ng = (t & 3) * 16;
      const int c = kk + k;
      const int h = c >> 6, d = c & 63;
      const _Float16* vrow =
          qkvh + ((size_t)b * QKV_ROWS + h * HROW + 1 + HDIM + d) * NSEQ + n0 + ng;
      if (kk + 32 < DIMC)
        __builtin_prefetch(vrow + (size_t)32 * NSEQ, 0, 3);  // global_prefetch_b8
      const float scale = ctx[((size_t)b * NHEAD + h) * HDIM + d];
      v8h h0 = *(const v8h*)(vrow);
      v8h h1 = *(const v8h*)(vrow + 8);
      _Float16* dst = Blds + (size_t)ng * LDB + k;
#pragma unroll
      for (int j = 0; j < 8; ++j)
        dst[j * LDB] = (_Float16)(fmaxf((float)h0[j], 0.f) * scale);
#pragma unroll
      for (int j = 0; j < 8; ++j)
        dst[(8 + j) * LDB] = (_Float16)(fmaxf((float)h1[j], 0.f) * scale);
    }
    __syncthreads();

    v16h a0 = fragA_g(Wph, m0 + wr,      kk, lane);
    v16h a1 = fragA_g(Wph, m0 + wr + 16, kk, lane);
    v16h b0 = fragB_lds(Blds, wc,      lane);
    v16h b1 = fragB_lds(Blds, wc + 16, lane);
    acc[0][0] = __builtin_amdgcn_wmma_f32_16x16x32_f16(false, a0, false, b0, (short)0, acc[0][0], false, false);
    acc[0][1] = __builtin_amdgcn_wmma_f32_16x16x32_f16(false, a0, false, b1, (short)0, acc[0][1], false, false);
    acc[1][0] = __builtin_amdgcn_wmma_f32_16x16x32_f16(false, a1, false, b0, (short)0, acc[1][0], false, false);
    acc[1][1] = __builtin_amdgcn_wmma_f32_16x16x32_f16(false, a1, false, b1, (short)0, acc[1][1], false, false);
  }

  const int g = lane >> 4, nl = lane & 15;
#pragma unroll
  for (int i = 0; i < 2; ++i)
#pragma unroll
    for (int j = 0; j < 2; ++j) {
      const int obase = m0 + wr + i * 16 + 8 * g;
      const int n     = n0 + wc + j * 16 + nl;
#pragma unroll
      for (int r = 0; r < 8; ++r) {
        const int o = obase + r;
        out[((size_t)b * DIMC + o) * NSEQ + n] = acc[i][j][r] + bias[o];
      }
    }
}

// ---------------------------------------------------------------------------
extern "C" void kernel_launch(void* const* d_in, const int* in_sizes, int n_in,
                              void* d_out, int out_size, void* d_ws, size_t ws_size,
                              hipStream_t stream) {
  const float* x    = (const float*)d_in[0];
  const float* Wqkv = (const float*)d_in[1];
  const float* bqkv = (const float*)d_in[2];
  const float* Wp   = (const float*)d_in[3];
  const float* bp   = (const float*)d_in[4];
  float* out = (float*)d_out;

  // workspace carve-up (all 256B aligned sizes)
  char* ws = (char*)d_ws;
  _Float16* qkvh = (_Float16*)ws;                 // 16*1032*4096*2 = 135.3 MB
  ws += (size_t)NBATCH * QKV_ROWS * NSEQ * sizeof(_Float16);
  _Float16* xTh = (_Float16*)ws;                  // 16*4096*512*2 = 67.1 MB
  ws += (size_t)NBATCH * NSEQ * DIMC * sizeof(_Float16);
  _Float16* Wh = (_Float16*)ws;                   // 1088*512*2
  ws += (size_t)QKV_PAD * DIMC * sizeof(_Float16);
  _Float16* Wph = (_Float16*)ws;                  // 512*512*2
  ws += (size_t)DIMC * DIMC * sizeof(_Float16);
  float* ctx = (float*)ws;                        // 16*8*64*4

  // k0a: weights -> f16
  wconv_kernel<<<(QKV_PAD * DIMC + 255) / 256, 256, 0, stream>>>(
      Wqkv, Wh, QKV_ROWS, QKV_PAD * DIMC);
  wconv_kernel<<<(DIMC * DIMC + 255) / 256, 256, 0, stream>>>(
      Wp, Wph, DIMC, DIMC * DIMC);

  // k0b: x -> xT f16 (TDM-staged tiles)
  x_transpose_kernel<<<dim3(NSEQ / 64, DIMC / 64, NBATCH), 256, 0, stream>>>(x, xTh);

  // k1: QKV GEMM (LDS-free WMMA)
  qkv_gemm_kernel<<<dim3(NSEQ / 128, QKV_PAD / 64, NBATCH), 128, 0, stream>>>(
      Wh, xTh, bqkv, qkvh);

  // k2: softmax + ctx
  softmax_ctx_kernel<<<dim3(NBATCH * NHEAD), 256, 0, stream>>>(qkvh, ctx);

  // k3: projection GEMM with fused relu(v)*ctx
  proj_gemm_kernel<<<dim3(NSEQ / 64, DIMC / 64, NBATCH), 128, 0, stream>>>(
      qkvh, ctx, Wph, bp, out);
}
